// MultiHeadedSelfAttention_21036749815913
// MI455X (gfx1250) — matrix operand order repaired and checked by reading, not compiled, not run on hardware
//
#include <hip/hip_runtime.h>

#define HIDDEN 1024
#define HEADS  16
#define DK     64
#define BATCH  4
#define SEQ    2048
#define M_TOTAL (BATCH * SEQ)   // 8192

typedef __attribute__((ext_vector_type(16))) __bf16 v16bf;
typedef __attribute__((ext_vector_type(8)))  float  v8f;

union FragBF {
    uint4 u[2];   // 32 bytes = 16 bf16 elements
    v16bf v;
};

__device__ __forceinline__ unsigned short f32_to_bf16_rne(float f) {
    unsigned int u = __float_as_uint(f);
    u += 0x7FFFu + ((u >> 16) & 1u);
    return (unsigned short)(u >> 16);
}

// ---------------------------------------------------------------------------
// TDM: 64x64 bf16 2-D tile, global -> LDS, per CDNA5 ISA
// ---------------------------------------------------------------------------
#if defined(__has_builtin)
#if __has_builtin(__builtin_amdgcn_tensor_load_to_lds)
#define HAVE_TDM 1
#endif
#endif

#ifdef HAVE_TDM
typedef __attribute__((ext_vector_type(4))) unsigned int u32x4;
typedef __attribute__((ext_vector_type(4))) int i32x4;
typedef __attribute__((ext_vector_type(8))) int i32x8;

__device__ __forceinline__ void tdm_load_tile_64x64(unsigned int lds_byte_off,
                                                    const unsigned short* gsrc) {
    const unsigned long long ga = (unsigned long long)(size_t)gsrc;
    // D# group 0: count=1 (user), lds_addr[63:32], global_addr[120:64], type=2
    const u32x4 g0 = {
        1u,
        lds_byte_off,
        (unsigned int)(ga & 0xFFFFFFFFull),
        (unsigned int)((ga >> 32) & 0x01FFFFFFull) | (2u << 30)
    };
    // D# group 1:
    //  w0: workgroup_mask=0, data_size=1 (2 bytes)
    //  w1[31:16] = tensor_dim0 (64)      w2[31:16] = tensor_dim1 lo (2048)
    //  w3[31:16] = tile_dim0 (64)        w4[15:0]  = tile_dim1 (64)
    //  w5        = tensor_dim0_stride (64 elements)
    const i32x8 g1 = {
        0x00010000,
        (int)(DK << 16),
        (int)(SEQ << 16),
        (int)(DK << 16),
        64,
        DK,
        0, 0
    };
    const i32x4 gz4 = {0, 0, 0, 0};
#if __clang_major__ >= 23
    const i32x8 gz8 = {0, 0, 0, 0, 0, 0, 0, 0};
    __builtin_amdgcn_tensor_load_to_lds(g0, g1, gz4, gz4, gz8, 0);
#else
    __builtin_amdgcn_tensor_load_to_lds(g0, g1, gz4, gz4, 0);
#endif
}
#endif // HAVE_TDM

// ---------------------------------------------------------------------------
// Kernel 1: fp32 -> bf16 (RNE), vectorized x4
// ---------------------------------------------------------------------------
__global__ void convert_f32_to_bf16(const float* __restrict__ src,
                                    unsigned short* __restrict__ dst, int n4) {
    int i = blockIdx.x * blockDim.x + threadIdx.x;
    int stride = gridDim.x * blockDim.x;
    for (; i < n4; i += stride) {
        float4 f = ((const float4*)src)[i];
        ushort4 r;
        r.x = f32_to_bf16_rne(f.x);
        r.y = f32_to_bf16_rne(f.y);
        r.z = f32_to_bf16_rne(f.z);
        r.w = f32_to_bf16_rne(f.w);
        ((ushort4*)dst)[i] = r;
    }
}

// ---------------------------------------------------------------------------
// Kernel 2: projection GEMM  C[m,n] = (sum_k X[m,k] * W[n,k] + bias[n]) * scale
//   Register-blocked: each wave computes a 32x64 macro-tile
//   (2 A-frags x 4 B-frags -> 8 v_wmma per 32-wide k-step).
// ---------------------------------------------------------------------------
__global__ __launch_bounds__(256)
void qkv_proj_kernel(const unsigned short* __restrict__ X,
                     const unsigned short* __restrict__ W,
                     const float* __restrict__ bias,
                     unsigned short* __restrict__ Out,
                     float scale) {
    const int lane   = threadIdx.x & 31;
    const int wave   = threadIdx.x >> 5;
    const int tile   = blockIdx.x * 8 + wave;   // 4096 wave-tiles total
    const int tileM  = tile >> 4;               // 1024/64 = 16 N macro-tiles
    const int tileN  = tile & 15;
    const int hiHalf = lane >> 4;

    const int row0  = tileM * 32;
    const int ncol0 = tileN * 64;

    const unsigned short* Arow[2];
    const unsigned short* Brow[4];
#pragma unroll
    for (int i = 0; i < 2; ++i)
        Arow[i] = X + (size_t)(row0 + i * 16 + (lane & 15)) * HIDDEN;
#pragma unroll
    for (int jn = 0; jn < 4; ++jn)
        Brow[jn] = W + (size_t)(ncol0 + jn * 16 + (lane & 15)) * HIDDEN;

    v8f acc[2][4] = {{{}, {}, {}, {}}, {{}, {}, {}, {}}};

    for (int k = 0; k < HIDDEN; k += 32) {
        FragBF a[2], b[4];
        const int ka = k + hiHalf * 8;    // A 16x32 layout
        const int kb = k + hiHalf * 8;    // B 32x16 layout (same k order as A)
#pragma unroll
        for (int i = 0; i < 2; ++i) {
            a[i].u[0] = *(const uint4*)(Arow[i] + ka);
            a[i].u[1] = *(const uint4*)(Arow[i] + ka + 16);
        }
#pragma unroll
        for (int jn = 0; jn < 4; ++jn) {
            b[jn].u[0] = *(const uint4*)(Brow[jn] + kb);
            b[jn].u[1] = *(const uint4*)(Brow[jn] + kb + 16);
        }
#pragma unroll
        for (int i = 0; i < 2; ++i)
#pragma unroll
            for (int jn = 0; jn < 4; ++jn)
                acc[i][jn] = __builtin_amdgcn_wmma_f32_16x16x32_bf16(
                    false, a[i].v, false, b[jn].v, (short)0, acc[i][jn],
                    false, false);
    }

#pragma unroll
    for (int jn = 0; jn < 4; ++jn) {
        const int n  = ncol0 + jn * 16 + (lane & 15);
        const float bn = bias[n];
        const int h = n / DK, d = n % DK;
#pragma unroll
        for (int i = 0; i < 2; ++i) {
#pragma unroll
            for (int j = 0; j < 8; ++j) {
                int m  = row0 + i * 16 + j + hiHalf * 8;  // C layout row mapping
                int bb = m / SEQ, s = m % SEQ;
                float v = (acc[i][jn][j] + bn) * scale;
                Out[(((size_t)(bb * HEADS + h) * SEQ) + s) * DK + d] =
                    f32_to_bf16_rne(v);
            }
        }
    }
}

// ---------------------------------------------------------------------------
// Kernel 3: flash attention.  One block = one (b,h) x 128 query rows.
//   8 waves, each owns 16 q-rows.  64-key KV tiles:
//     K tile: TDM (tensor_load_to_lds) straight into LDS, TENSORcnt-waited
//     V tile: manually transposed into LDS (TDM cannot transpose)
// ---------------------------------------------------------------------------
__global__ __launch_bounds__(256)
void flash_attn_kernel(const unsigned short* __restrict__ Q,   // [B,H,S,DK] bf16
                       const unsigned short* __restrict__ K,
                       const unsigned short* __restrict__ V,
                       unsigned short* __restrict__ Ctx)       // [B,S,HIDDEN] bf16
{
    __shared__ __align__(16) unsigned short Kt[64][DK];     // 64 keys x 64 dims
    __shared__ __align__(16) unsigned short Vt[DK][64];     // transposed: dim x key
    __shared__ __align__(16) unsigned short Pl[8][16][64];  // per-wave P tile

    const int lane   = threadIdx.x & 31;
    const int wave   = threadIdx.x >> 5;
    const int hiHalf = lane >> 4;
    const int qt     = blockIdx.x & 15;      // 16 q-tiles of 128 rows
    const int bh     = blockIdx.x >> 4;      // b*HEADS + h
    const int bb     = bh >> 4;
    const int h      = bh & 15;

    const unsigned short* Qb = Q + (size_t)bh * SEQ * DK;
    const unsigned short* Kb = K + (size_t)bh * SEQ * DK;
    const unsigned short* Vb = V + (size_t)bh * SEQ * DK;

    const int qrow = qt * 128 + wave * 16 + (lane & 15);

    // Q A-fragments for dk k-steps 0 and 32 (Q already has 1/sqrt(dk) applied)
    FragBF qa[2];
#pragma unroll
    for (int ks = 0; ks < 2; ++ks) {
        int k0 = ks * 32 + hiHalf * 8;
        qa[ks].u[0] = *(const uint4*)(Qb + (size_t)qrow * DK + k0);
        qa[ks].u[1] = *(const uint4*)(Qb + (size_t)qrow * DK + k0 + 16);
    }

    v8f acc[4] = {{}, {}, {}, {}};   // ctx accumulators: 4 x (16 dims)
    float mrow[8], lrow[8];
#pragma unroll
    for (int j = 0; j < 8; ++j) { mrow[j] = -3.0e38f; lrow[j] = 0.0f; }

#ifdef HAVE_TDM
    const unsigned int kt_lds = (unsigned int)(size_t)(&Kt[0][0]);
#endif

    for (int kv = 0; kv < SEQ; kv += 64) {
        // ---- K tile: TDM DMA straight into LDS (one issuing wave) ----
#ifdef HAVE_TDM
        if (wave == 0)
            tdm_load_tile_64x64(kt_lds, Kb + (size_t)kv * DK);
#else
#pragma unroll
        for (int rep = 0; rep < 2; ++rep) {
            int idx = threadIdx.x + rep * 256;
            int key = idx >> 3;            // 0..63
            int db  = (idx & 7) * 8;       // 0..56
            *(uint4*)(&Kt[key][db]) =
                *(const uint4*)(Kb + (size_t)(kv + key) * DK + db);
        }
#endif
        // ---- V tile: cooperative load + transpose into LDS ----
#pragma unroll
        for (int rep = 0; rep < 2; ++rep) {
            int idx = threadIdx.x + rep * 256;   // 512 uint4 = 64x64 tile
            int key = idx >> 3;
            int db  = (idx & 7) * 8;
            uint4 vvec = *(const uint4*)(Vb + (size_t)(kv + key) * DK + db);
            const unsigned short* vs = (const unsigned short*)&vvec;
#pragma unroll
            for (int i = 0; i < 8; ++i) Vt[db + i][key] = vs[i];
        }
#ifdef HAVE_TDM
        if (wave == 0) __builtin_amdgcn_s_wait_tensorcnt(0);
#endif
        __syncthreads();

        // ---- scores: S = Q @ K^T for four 16-key subtiles ----
        v8f sc[4];
#pragma unroll
        for (int sub = 0; sub < 4; ++sub) {
            v8f c = {};
#pragma unroll
            for (int ks = 0; ks < 2; ++ks) {
                FragBF bfr;
                int col = sub * 16 + (lane & 15);    // key column
                int kb2 = ks * 32 + hiHalf * 8;      // dk offset
                bfr.u[0] = *(const uint4*)(&Kt[col][kb2]);
                bfr.u[1] = *(const uint4*)(&Kt[col][kb2 + 16]);
                c = __builtin_amdgcn_wmma_f32_16x16x32_bf16(false, qa[ks].v, false,
                                                            bfr.v, (short)0, c,
                                                            false, false);
            }
            sc[sub] = c;
        }

        // ---- online softmax (fp32), width-16 row reductions ----
#pragma unroll
        for (int j = 0; j < 8; ++j) {
            float tmax = fmaxf(fmaxf(sc[0][j], sc[1][j]),
                               fmaxf(sc[2][j], sc[3][j]));
#pragma unroll
            for (int off = 1; off < 16; off <<= 1)
                tmax = fmaxf(tmax, __shfl_xor(tmax, off, 16));
            float mn   = fmaxf(mrow[j], tmax);
            float corr = __expf(mrow[j] - mn);
            float p[4];
            float rs = 0.0f;
#pragma unroll
            for (int sub = 0; sub < 4; ++sub) {
                p[sub] = __expf(sc[sub][j] - mn);
                rs += p[sub];
            }
#pragma unroll
            for (int off = 1; off < 16; off <<= 1)
                rs += __shfl_xor(rs, off, 16);
            lrow[j] = lrow[j] * corr + rs;
            mrow[j] = mn;
#pragma unroll
            for (int n = 0; n < 4; ++n) acc[n][j] *= corr;
            // stash P into per-wave LDS (C-layout -> A-layout round trip)
#pragma unroll
            for (int sub = 0; sub < 4; ++sub)
                Pl[wave][j + hiHalf * 8][sub * 16 + (lane & 15)] =
                    f32_to_bf16_rne(p[sub]);
        }
        __syncthreads();   // orders LDS stores vs. cross-lane loads

        // P A-fragments over the 64-key K-dim (two 32-wide k-steps)
        FragBF pa[2];
#pragma unroll
        for (int ks = 0; ks < 2; ++ks) {
            int r  = lane & 15;
            int k0 = ks * 32 + hiHalf * 8;
            pa[ks].u[0] = *(const uint4*)(&Pl[wave][r][k0]);
            pa[ks].u[1] = *(const uint4*)(&Pl[wave][r][k0 + 16]);
        }

        // ---- ctx += P @ V (B-fragments contiguous from transposed V) ----
#pragma unroll
        for (int n = 0; n < 4; ++n) {
            int dcol = n * 16 + (lane & 15);   // feature column
#pragma unroll
            for (int ks = 0; ks < 2; ++ks) {
                FragBF bv;
                int kb2 = ks * 32 + hiHalf * 8;    // key offset
                bv.u[0] = *(const uint4*)(&Vt[dcol][kb2]);
                bv.u[1] = *(const uint4*)(&Vt[dcol][kb2 + 16]);
                acc[n] = __builtin_amdgcn_wmma_f32_16x16x32_bf16(
                    false, pa[ks].v, false, bv.v, (short)0, acc[n], false, false);
            }
        }
        __syncthreads();   // protect Kt/Vt before next tile load
    }

    // ---- epilogue: ctx = acc / l, write [B,S,HIDDEN] with head-major cols ----
#pragma unroll
    for (int n = 0; n < 4; ++n) {
        int dcol = n * 16 + (lane & 15);
#pragma unroll
        for (int j = 0; j < 8; ++j) {
            int r = qt * 128 + wave * 16 + j + hiHalf * 8;
            float v = acc[n][j] / lrow[j];
            Ctx[((size_t)(bb * SEQ + r)) * HIDDEN + h * DK + dcol] = f32_to_bf16_rne(v);
        }
    }
}

// ---------------------------------------------------------------------------
// Kernel 4: output projection, register-blocked 32x64 per wave, fp32 out + bias
// ---------------------------------------------------------------------------
__global__ __launch_bounds__(256)
void out_proj_kernel(const unsigned short* __restrict__ X,   // ctx bf16 [8192,1024]
                     const unsigned short* __restrict__ W,   // Wo bf16 [1024,1024]
                     const float* __restrict__ bias,
                     float* __restrict__ Out) {
    const int lane   = threadIdx.x & 31;
    const int wave   = threadIdx.x >> 5;
    const int tile   = blockIdx.x * 8 + wave;
    const int tileM  = tile >> 4;
    const int tileN  = tile & 15;
    const int hiHalf = lane >> 4;

    const int row0  = tileM * 32;
    const int ncol0 = tileN * 64;

    const unsigned short* Arow[2];
    const unsigned short* Brow[4];
#pragma unroll
    for (int i = 0; i < 2; ++i)
        Arow[i] = X + (size_t)(row0 + i * 16 + (lane & 15)) * HIDDEN;
#pragma unroll
    for (int jn = 0; jn < 4; ++jn)
        Brow[jn] = W + (size_t)(ncol0 + jn * 16 + (lane & 15)) * HIDDEN;

    v8f acc[2][4] = {{{}, {}, {}, {}}, {{}, {}, {}, {}}};

    for (int k = 0; k < HIDDEN; k += 32) {
        FragBF a[2], b[4];
        const int ka = k + hiHalf * 8;
        const int kb = k + hiHalf * 8;
#pragma unroll
        for (int i = 0; i < 2; ++i) {
            a[i].u[0] = *(const uint4*)(Arow[i] + ka);
            a[i].u[1] = *(const uint4*)(Arow[i] + ka + 16);
        }
#pragma unroll
        for (int jn = 0; jn < 4; ++jn) {
            b[jn].u[0] = *(const uint4*)(Brow[jn] + kb);
            b[jn].u[1] = *(const uint4*)(Brow[jn] + kb + 16);
        }
#pragma unroll
        for (int i = 0; i < 2; ++i)
#pragma unroll
            for (int jn = 0; jn < 4; ++jn)
                acc[i][jn] = __builtin_amdgcn_wmma_f32_16x16x32_bf16(
                    false, a[i].v, false, b[jn].v, (short)0, acc[i][jn],
                    false, false);
    }

#pragma unroll
    for (int jn = 0; jn < 4; ++jn) {
        const int n  = ncol0 + jn * 16 + (lane & 15);
        const float bn = bias[n];
#pragma unroll
        for (int i = 0; i < 2; ++i) {
#pragma unroll
            for (int j = 0; j < 8; ++j) {
                int m = row0 + i * 16 + j + hiHalf * 8;
                Out[(size_t)m * HIDDEN + n] = acc[i][jn][j] + bn;
            }
        }
    }
}

// ---------------------------------------------------------------------------
extern "C" void kernel_launch(void* const* d_in, const int* in_sizes, int n_in,
                              void* d_out, int out_size, void* d_ws, size_t ws_size,
                              hipStream_t stream) {
    (void)in_sizes; (void)n_in; (void)out_size; (void)ws_size;

    const float* tgt = (const float*)d_in[0];
    const float* Wq  = (const float*)d_in[1];
    const float* bq  = (const float*)d_in[2];
    const float* Wk  = (const float*)d_in[3];
    const float* bk  = (const float*)d_in[4];
    const float* Wv  = (const float*)d_in[5];
    const float* bv  = (const float*)d_in[6];
    const float* Wo  = (const float*)d_in[7];
    const float* bo  = (const float*)d_in[8];
    float* out = (float*)d_out;

    unsigned short* ws = (unsigned short*)d_ws;
    size_t o = 0;
    unsigned short* tgt_bf = ws + o; o += (size_t)M_TOTAL * HIDDEN;
    unsigned short* wq_bf  = ws + o; o += (size_t)HIDDEN * HIDDEN;
    unsigned short* wk_bf  = ws + o; o += (size_t)HIDDEN * HIDDEN;
    unsigned short* wv_bf  = ws + o; o += (size_t)HIDDEN * HIDDEN;
    unsigned short* wo_bf  = ws + o; o += (size_t)HIDDEN * HIDDEN;
    unsigned short* Qw     = ws + o; o += (size_t)M_TOTAL * HIDDEN;  // [B,H,S,DK]
    unsigned short* Kw     = ws + o; o += (size_t)M_TOTAL * HIDDEN;
    unsigned short* Vw     = ws + o; o += (size_t)M_TOTAL * HIDDEN;
    unsigned short* ctx_bf = ws + o; o += (size_t)M_TOTAL * HIDDEN;  // [B,S,HIDDEN]

    // 1) fp32 -> bf16 conversions
    {
        int n4 = (M_TOTAL * HIDDEN) / 4;
        convert_f32_to_bf16<<<2048, 256, 0, stream>>>(tgt, tgt_bf, n4);
        int w4 = (HIDDEN * HIDDEN) / 4;
        convert_f32_to_bf16<<<1024, 256, 0, stream>>>(Wq, wq_bf, w4);
        convert_f32_to_bf16<<<1024, 256, 0, stream>>>(Wk, wk_bf, w4);
        convert_f32_to_bf16<<<1024, 256, 0, stream>>>(Wv, wv_bf, w4);
        convert_f32_to_bf16<<<1024, 256, 0, stream>>>(Wo, wo_bf, w4);
    }

    // 2) Q/K/V projections: 256x16 = 4096 wave-tiles, 8 waves/block -> 512 blocks
    qkv_proj_kernel<<<512, 256, 0, stream>>>(tgt_bf, wq_bf, bq, Qw, 0.125f); // 1/sqrt(64)
    qkv_proj_kernel<<<512, 256, 0, stream>>>(tgt_bf, wk_bf, bk, Kw, 1.0f);
    qkv_proj_kernel<<<512, 256, 0, stream>>>(tgt_bf, wv_bf, bv, Vw, 1.0f);

    // 3) flash attention: B*HEADS*(SEQ/128) = 1024 blocks
    flash_attn_kernel<<<BATCH * HEADS * (SEQ / 128), 256, 0, stream>>>(Qw, Kw, Vw, ctx_bf);

    // 4) output projection
    out_proj_kernel<<<512, 256, 0, stream>>>(ctx_bf, wo_bf, bo, out);
}
